// SparseDeformableAttentionBlockWithRoPE_29489245454357
// MI455X (gfx1250) — matrix operand order repaired and checked by reading, not compiled, not run on hardware
//
#include <hip/hip_runtime.h>

// ---------- types for WMMA ----------
typedef __attribute__((ext_vector_type(16))) __bf16       v16bf;
typedef __attribute__((ext_vector_type(8)))  float        v8f;
typedef __attribute__((ext_vector_type(8)))  unsigned int v8u;
typedef __attribute__((ext_vector_type(4)))  unsigned int v4u;

// ---------- constants ----------
#define BB   2
#define NQ   8192
#define DD   256
#define NHH  8
#define HD_  32
#define NLL  4
#define NPP  4
#define LV   13294          // 100*100 + 50*50 + 25*25 + 13*13
#define NTOK (BB*NQ)        // 16384
#define MV   (BB*LV)        // 26588

__device__ __forceinline__ unsigned short f32_to_bf16(float f) {
    unsigned int u = __float_as_uint(f);
    u += 0x7FFFu + ((u >> 16) & 1u);          // round-to-nearest-even
    return (unsigned short)(u >> 16);
}

// ---------- weight transpose + bf16 convert: Wt[n*K + k] = W[k*N + n] ----------
__global__ void transpose_bf16_kernel(const float* __restrict__ W,
                                      unsigned short* __restrict__ Wt,
                                      int K, int N) {
    int idx = blockIdx.x * blockDim.x + threadIdx.x;
    if (idx >= K * N) return;
    int k = idx / N, n = idx % N;
    Wt[(size_t)n * K + k] = f32_to_bf16(W[(size_t)k * N + n]);
}

// ---------- elementwise f32 -> bf16 (for the value tensor) ----------
__global__ void convert_bf16_kernel(const float* __restrict__ src,
                                    unsigned short* __restrict__ dst, int n) {
    int idx = blockIdx.x * blockDim.x + threadIdx.x;
    if (idx < n) dst[idx] = f32_to_bf16(src[idx]);
}

// ---------- LayerNorm + 2D RoPE (one 128-thread block per token), bf16 output ----------
__global__ __launch_bounds__(128)
void ln_rope_kernel(const float* __restrict__ x, const float* __restrict__ qpos,
                    const float* __restrict__ g, const float* __restrict__ bln,
                    unsigned short* __restrict__ hr) {
    int token = blockIdx.x;                   // 0..NTOK-1
    int t = threadIdx.x;                      // 0..127 : pair index
    const float* xr = x + (size_t)token * DD;
    float a = xr[2 * t], b = xr[2 * t + 1];
    float s = a + b, ss = a * a + b * b;
    #pragma unroll
    for (int off = 16; off > 0; off >>= 1) {
        s  += __shfl_xor(s,  off);
        ss += __shfl_xor(ss, off);
    }
    __shared__ float sh[8];
    int wid = t >> 5;
    if ((t & 31) == 0) { sh[wid * 2] = s; sh[wid * 2 + 1] = ss; }
    __syncthreads();
    float S  = sh[0] + sh[2] + sh[4] + sh[6];
    float SS = sh[1] + sh[3] + sh[5] + sh[7];
    float mean = S * (1.0f / 256.0f);
    float var  = SS * (1.0f / 256.0f) - mean * mean;
    float rstd = rsqrtf(var + 1e-5f);
    float h1 = (a - mean) * rstd * g[2 * t]     + bln[2 * t];
    float h2 = (b - mean) * rstd * g[2 * t + 1] + bln[2 * t + 1];
    // RoPE: pair k -> angle from pos[x] (k<64) or pos[y] (k>=64), freq i = k%64
    int i = t & 63;
    float p   = qpos[(size_t)token * 2 + (t >> 6)];
    float ang = p * 100.0f * exp2f(-(float)i * (13.287712379549449f / 64.0f));
    float c = cosf(ang), sn = sinf(ang);
    unsigned short* o = hr + (size_t)token * DD;
    o[2 * t]     = f32_to_bf16(h1 * c - h2 * sn);
    o[2 * t + 1] = f32_to_bf16(h1 * sn + h2 * c);
}

// ---------- bf16 WMMA GEMM: C[M,N] = A[M,256] * Wt[N,256]^T + bias (+residual) ----------
// A row-major bf16; Wt N-major bf16 (contiguous along K).
// Each wave computes a 16(M) x 64(N) slab: one A fragment feeds 4 WMMAs per k-step.
__global__ __launch_bounds__(128)
void wmma_gemm_kernel(const unsigned short* __restrict__ A,
                      const unsigned short* __restrict__ Wt,
                      const float* __restrict__ bias, const float* __restrict__ residual,
                      float* __restrict__ Cout, int M, int N) {
    const int K = 256;
    int lane = threadIdx.x & 31;
    int task = blockIdx.x * 4 + (threadIdx.x >> 5);     // one 16x64 slab per wave
    int groups = N >> 6;                                // column-groups of 64
    int totalTasks = ((M + 15) >> 4) * groups;
    if (task >= totalTasks) return;                     // wave-uniform
    int tileM = task / groups, gN = task % groups;
    int m  = lane & 15;
    int hi = lane >> 4;

    int rowA = tileM * 16 + m;
    if (rowA > M - 1) rowA = M - 1;                     // clamp; discarded at store
    // A fragment: lane half selects K sub-range; two contiguous 16B runs per k-step
    const unsigned short* arow = A + (size_t)rowA * K + hi * 8;
    // B fragments: lane holds column n = lane&15 (+16*j per subtile); two 16B runs per k-step, same K order as A
    const unsigned short* bcol = Wt + (size_t)(gN * 64 + m) * K + hi * 8;

    v8f acc[4] = {v8f{}, v8f{}, v8f{}, v8f{}};
    #pragma unroll
    for (int k0 = 0; k0 < K; k0 += 32) {
        v4u alo = *(const v4u*)(arow + k0);          // K pairs 0..7   (lane-half offset)
        v4u ahi = *(const v4u*)(arow + k0 + 16);     // K pairs 16..23 (lane-half offset)
        v8u au;
        #pragma unroll
        for (int q = 0; q < 4; ++q) { au[q] = alo[q]; au[q + 4] = ahi[q]; }
        v16bf av = __builtin_bit_cast(v16bf, au);
        #pragma unroll
        for (int j = 0; j < 4; ++j) {
            v4u blo = *(const v4u*)(bcol + (size_t)j * 16 * K + k0);
            v4u bhi = *(const v4u*)(bcol + (size_t)j * 16 * K + k0 + 16);
            v8u bu;
            #pragma unroll
            for (int q = 0; q < 4; ++q) { bu[q] = blo[q]; bu[q + 4] = bhi[q]; }
            acc[j] = __builtin_amdgcn_wmma_f32_16x16x32_bf16(
                false, av, false, __builtin_bit_cast(v16bf, bu),
                (short)0, acc[j], false, false);
        }
    }
    // C/D layout: element r of v8f = row (r + 8*hi), col = lane&15
    #pragma unroll
    for (int j = 0; j < 4; ++j) {
        int col = gN * 64 + j * 16 + (lane & 15);
        float bc = bias ? bias[col] : 0.0f;
        #pragma unroll
        for (int r = 0; r < 8; ++r) {
            int rowOut = tileM * 16 + r + hi * 8;
            if (rowOut < M) {
                float v = acc[j][r] + bc;
                if (residual) v += residual[(size_t)rowOut * N + col];
                Cout[(size_t)rowOut * N + col] = v;
            }
        }
    }
}

// ---------- per-head softmax over 16 logits (in place) ----------
__global__ void softmax16_kernel(float* __restrict__ attn, int total) {
    int idx = blockIdx.x * blockDim.x + threadIdx.x;
    if (idx >= total) return;                 // total = NTOK*NH
    int token = idx / NHH, h = idx % NHH;
    float* p = attn + (size_t)token * (NHH * NLL * NPP) + h * (NLL * NPP);
    float mx = -1e30f;
    float v[16];
    #pragma unroll
    for (int j = 0; j < 16; ++j) { v[j] = p[j]; mx = fmaxf(mx, v[j]); }
    float s = 0.0f;
    #pragma unroll
    for (int j = 0; j < 16; ++j) { v[j] = __expf(v[j] - mx); s += v[j]; }
    float inv = 1.0f / s;
    #pragma unroll
    for (int j = 0; j < 16; ++j) p[j] = v[j] * inv;
}

// ---------- deformable bilinear sampling: one wave per (b,n,h), lane = channel ----------
__global__ __launch_bounds__(128)
void sample_kernel(const float* __restrict__ vproj, const float* __restrict__ refp,
                   const float* __restrict__ offb, const float* __restrict__ attn,
                   unsigned short* __restrict__ samp) {
    int g    = (blockIdx.x * blockDim.x + threadIdx.x) >> 5;   // wave id, 0..NTOK*NH-1
    int lane = threadIdx.x & 31;                               // channel d
    int b   = g / (NQ * NHH);
    int rem = g - b * (NQ * NHH);
    int n   = rem / NHH;
    int h   = rem - n * NHH;
    size_t tok = (size_t)b * NQ + n;

    const int Hs[4] = {100, 50, 25, 13};
    const int Wl[4] = {100, 50, 25, 13};
    const int LS[4] = {0, 10000, 12500, 13125};

    const float* vbase = vproj + (size_t)b * LV * DD + h * HD_ + lane;
    float acc = 0.0f;
    #pragma unroll
    for (int l = 0; l < 4; ++l) {
        int   Wi = Wl[l], Hi = Hs[l], base = LS[l];
        float Wf = (float)Wi, Hf = (float)Hi;
        float refx = refp[(tok * NLL + l) * 2 + 0];
        float refy = refp[(tok * NLL + l) * 2 + 1];
        #pragma unroll
        for (int p = 0; p < 4; ++p) {
            int oc = ((h * NLL + l) * NPP + p) * 2;
            float offx = offb[tok * DD + oc];
            float offy = offb[tok * DD + oc + 1];
            float aw   = attn[tok * (NHH * NLL * NPP) + (h * NLL + l) * NPP + p];
            float px = (refx + offx / Wf) * Wf - 0.5f;
            float py = (refy + offy / Hf) * Hf - 0.5f;
            float x0f = floorf(px), y0f = floorf(py);
            float dx = px - x0f, dy = py - y0f;
            int x0 = (int)x0f, y0 = (int)y0f;
            float wts[4] = {(1.f - dx) * (1.f - dy), dx * (1.f - dy),
                            (1.f - dx) * dy,          dx * dy};
            int xs[4] = {x0, x0 + 1, x0,     x0 + 1};
            int ys[4] = {y0, y0,     y0 + 1, y0 + 1};
            #pragma unroll
            for (int c = 0; c < 4; ++c) {
                int xi = xs[c], yi = ys[c];
                if (xi >= 0 && xi < Wi && yi >= 0 && yi < Hi) {  // wave-uniform branch
                    float val = vbase[(size_t)(base + yi * Wi + xi) * DD];
                    acc += aw * wts[c] * val;
                }
            }
        }
    }
    samp[tok * DD + h * HD_ + lane] = f32_to_bf16(acc);
}

// ---------- host launch ----------
extern "C" void kernel_launch(void* const* d_in, const int* in_sizes, int n_in,
                              void* d_out, int out_size, void* d_ws, size_t ws_size,
                              hipStream_t stream) {
    const float* x     = (const float*)d_in[0];
    const float* qpos  = (const float*)d_in[1];
    const float* refp  = (const float*)d_in[2];
    const float* value = (const float*)d_in[3];
    const float* ln_g  = (const float*)d_in[4];
    const float* ln_b  = (const float*)d_in[5];
    const float* Wv    = (const float*)d_in[6];
    const float* bv    = (const float*)d_in[7];
    const float* Woff  = (const float*)d_in[8];
    const float* boff  = (const float*)d_in[9];
    const float* Wattn = (const float*)d_in[10];
    const float* battn = (const float*)d_in[11];
    const float* Wout  = (const float*)d_in[12];
    const float* bout  = (const float*)d_in[13];
    float* out = (float*)d_out;

    char* ws = (char*)d_ws;
    size_t off = 0;
    auto alloc = [&](size_t bytes) -> char* {
        char* p = ws + off;
        off += (bytes + 255) & ~(size_t)255;
        return p;
    };
    unsigned short* WvT    = (unsigned short*)alloc((size_t)256 * 256 * 2);
    unsigned short* WoffT  = (unsigned short*)alloc((size_t)256 * 256 * 2);
    unsigned short* WattnT = (unsigned short*)alloc((size_t)128 * 256 * 2);
    unsigned short* WoutT  = (unsigned short*)alloc((size_t)256 * 256 * 2);
    unsigned short* hr     = (unsigned short*)alloc((size_t)NTOK * DD * 2);
    unsigned short* valbf  = (unsigned short*)alloc((size_t)MV * DD * 2);
    unsigned short* samp   = (unsigned short*)alloc((size_t)NTOK * DD * 2);
    float* vproj = (float*)alloc((size_t)MV * DD * 4);
    float* offb  = (float*)alloc((size_t)NTOK * DD * 4);
    float* attnb = (float*)alloc((size_t)NTOK * 128 * 4);

    // weights -> bf16, N-major (contiguous along K for WMMA B fragments)
    transpose_bf16_kernel<<<(256 * 256 + 255) / 256, 256, 0, stream>>>(Wv,    WvT,    256, 256);
    transpose_bf16_kernel<<<(256 * 256 + 255) / 256, 256, 0, stream>>>(Woff,  WoffT,  256, 256);
    transpose_bf16_kernel<<<(256 * 128 + 255) / 256, 256, 0, stream>>>(Wattn, WattnT, 256, 128);
    transpose_bf16_kernel<<<(256 * 256 + 255) / 256, 256, 0, stream>>>(Wout,  WoutT,  256, 256);

    // value -> bf16 (one-time elementwise pass)
    convert_bf16_kernel<<<((MV * DD) + 255) / 256, 256, 0, stream>>>(value, valbf, MV * DD);

    // LN + RoPE, bf16 output
    ln_rope_kernel<<<NTOK, 128, 0, stream>>>(x, qpos, ln_g, ln_b, hr);

    // value projection (26588 x 256 x 256): tasks = ceil(M/16) * (N/64)
    int tasks_v = ((MV + 15) / 16) * (256 / 64);
    wmma_gemm_kernel<<<(tasks_v + 3) / 4, 128, 0, stream>>>(valbf, WvT, bv, nullptr, vproj, MV, 256);

    // offsets (16384 x 256 x 256) and attention logits (16384 x 128 x 256)
    int tasks_off  = (NTOK / 16) * (256 / 64);
    int tasks_attn = (NTOK / 16) * (128 / 64);
    wmma_gemm_kernel<<<tasks_off  / 4, 128, 0, stream>>>(hr, WoffT,  boff,  nullptr, offb,  NTOK, 256);
    wmma_gemm_kernel<<<tasks_attn / 4, 128, 0, stream>>>(hr, WattnT, battn, nullptr, attnb, NTOK, 128);

    // softmax over (NL*NP)=16 per head
    softmax16_kernel<<<(NTOK * NHH + 127) / 128, 128, 0, stream>>>(attnb, NTOK * NHH);

    // bilinear deformable sampling (L2-resident vproj), bf16 output
    sample_kernel<<<(NTOK * NHH) / 4, 128, 0, stream>>>(vproj, refp, offb, attnb, samp);

    // output projection + bias + residual
    wmma_gemm_kernel<<<tasks_off / 4, 128, 0, stream>>>(samp, WoutT, bout, x, out, NTOK, 256);
}
